// MutationToBlockPool_64355789963654
// MI455X (gfx1250) — matrix operand order repaired and checked by reading, not compiled, not run on hardware
//
#include <hip/hip_runtime.h>
#include <hip/hip_bf16.h>
#include <limits.h>

typedef __attribute__((ext_vector_type(16))) _Float16 v16h;
typedef __attribute__((ext_vector_type(8)))  _Float16 v8h;
typedef __attribute__((ext_vector_type(4)))  _Float16 v4h;
typedef __attribute__((ext_vector_type(8)))  float    v8f;

#define B_    8
#define N_    4096
#define H_    1024
#define HH_   512     // H/2
#define NB_   128
#define LN_EPS 1e-5f

typedef __attribute__((address_space(1))) int  gint;
typedef __attribute__((address_space(3))) int  sint;
typedef __attribute__((address_space(1))) char gchar;
typedef __attribute__((address_space(3))) char schar;

#if defined(__has_builtin)
#if __has_builtin(__builtin_amdgcn_global_load_async_to_lds_b32) && \
    __has_builtin(__builtin_amdgcn_global_load_async_to_lds_b8)
#define HAVE_ASYNC_LDS 1
#endif
#endif

__device__ __forceinline__ void async_wait0() {
#if defined(__has_builtin) && __has_builtin(__builtin_amdgcn_s_wait_asynccnt)
    __builtin_amdgcn_s_wait_asynccnt(0);
#else
    asm volatile("s_wait_asynccnt 0" ::: "memory");
#endif
}

// order-preserving float<->int encoding for integer atomicMax on floats
__device__ __forceinline__ int enc_f32(float f) {
    int i = __float_as_int(f);
    return i ^ ((i >> 31) & 0x7fffffff);
}
__device__ __forceinline__ float dec_f32(int k) {
    int i = k ^ ((k >> 31) & 0x7fffffff);
    return __int_as_float(i);
}
// branch-free tanh: (e^{2v}-1)/(e^{2v}+1) via v_exp_f32 + v_rcp_f32
__device__ __forceinline__ float fast_tanh(float v) {
    float e = __expf(2.f * v);
    return (e - 1.f) * __builtin_amdgcn_rcpf(e + 1.f);
}
__device__ __forceinline__ float eluf(float v) {
    return v > 0.f ? v : (__expf(v) - 1.f);
}

// ---------------------------------------------------------------------------
// Kernel 0: f32 -> f16 weight pre-conversion (W1, Wp) into workspace.
// ---------------------------------------------------------------------------
__global__ void __launch_bounds__(256) cvt_f16_kernel(
    const float* __restrict__ src, _Float16* __restrict__ dst, int n4)
{
    int i = blockIdx.x * 256 + threadIdx.x;
    if (i < n4) {
        float4 v = ((const float4*)src)[i];
        v4h h; h[0] = (_Float16)v.x; h[1] = (_Float16)v.y;
               h[2] = (_Float16)v.z; h[3] = (_Float16)v.w;
        *(v4h*)&dst[i * 4] = h;
    }
}

// ---------------------------------------------------------------------------
// Kernel 1: gate[b,n] = W2 . tanh(x @ W1 + b1) + b2   (fused WMMA GEMM)
// 256 WGs x 256 threads (8 waves); 128 tokens/WG staged f16 in 256KB LDS.
// ---------------------------------------------------------------------------
__global__ void __launch_bounds__(256) gate_kernel(
    const float* __restrict__ x, const _Float16* __restrict__ w1h,
    const float* __restrict__ b1, const float* __restrict__ W2,
    const float* __restrict__ b2, float* __restrict__ gate)
{
    extern __shared__ char smem[];
    _Float16* xs  = (_Float16*)smem;                       // 128*1024 f16
    float*    sb1 = (float*)(smem + 128 * H_ * sizeof(_Float16));
    float*    sw2 = sb1 + HH_;

    const int tid  = threadIdx.x;
    const int lane = tid & 31;
    const int w    = tid >> 5;
    const int tokBase = blockIdx.x * 128;

    // stage x tile f32 -> f16 (tile is contiguous in memory)
    const float4* xg = (const float4*)x + (size_t)tokBase * (H_ / 4);
    #pragma unroll 4
    for (int i = 0; i < 128; ++i) {
        int e4 = tid + i * 256;
        float4 v = xg[e4];
        v4h h; h[0] = (_Float16)v.x; h[1] = (_Float16)v.y;
               h[2] = (_Float16)v.z; h[3] = (_Float16)v.w;
        *(v4h*)&xs[e4 * 4] = h;
    }
    for (int i = tid; i < HH_; i += 256) { sb1[i] = b1[i]; sw2[i] = W2[i]; }
    __syncthreads();
    const float b2v = b2[0];

    float gsum[8];
    #pragma unroll
    for (int r = 0; r < 8; ++r) gsum[r] = 0.f;

    const int arow = lane & 15;          // A row (M) for this lane
    const int ahi  = (lane >> 4) & 1;    // lanes 16-31 hold k+8 groups
    const _Float16* xrow = &xs[(w * 16 + arow) * H_];

    for (int nt = 0; nt < 32; ++nt) {    // 512 output cols in tiles of 16
        __builtin_prefetch(w1h + (size_t)lane * HH_ + (((nt + 1) & 31) * 16), 0, 1);
        v8f c = {};
        #pragma unroll 4
        for (int kt = 0; kt < 32; ++kt) { // K = 1024 in steps of 32
            // A: 16x32 f16 from LDS (two contiguous 16B segments per lane)
            const _Float16* ap = xrow + kt * 32 + ahi * 8;
            v8h a0 = *(const v8h*)(ap);
            v8h a1 = *(const v8h*)(ap + 16);
            v16h a;
            #pragma unroll
            for (int e = 0; e < 8; ++e) { a[e] = a0[e]; a[e + 8] = a1[e]; }
            // B: lane L holds W1h col (nt*16+L%16), k in table order matching A
            v16h bm;
            #pragma unroll
            for (int e = 0; e < 8; ++e) {
                bm[e]     = w1h[(size_t)(kt * 32 + ahi * 8 + e) * HH_ + nt * 16 + arow];
                bm[e + 8] = w1h[(size_t)(kt * 32 + 16 + ahi * 8 + e) * HH_ + nt * 16 + arow];
            }
            c = __builtin_amdgcn_wmma_f32_16x16x32_f16(
                    false, a, false, bm, (short)0, c, false, false);
        }
        // epilogue: gate += sum_cols tanh(c + b1[col]) * W2[col]
        const int col0 = nt * 16 + (lane & 15);
        #pragma unroll
        for (int r = 0; r < 8; ++r) {
            float t = fast_tanh(c[r] + sb1[col0]) * sw2[col0];
            #pragma unroll
            for (int s = 1; s < 16; s <<= 1) t += __shfl_xor(t, s, 32);
            gsum[r] += t;
        }
    }
    if ((lane & 15) == 0) {
        #pragma unroll
        for (int r = 0; r < 8; ++r) {
            int token = tokBase + w * 16 + r + ahi * 8;
            gate[token] = gsum[r] + b2v;
        }
    }
}

// ---------------------------------------------------------------------------
// Kernel 2: segmented softmax stats per (b, block): max, sum(exp), count.
// Staging via GLOBAL_LOAD_ASYNC_TO_LDS (ASYNCcnt) when available.
// ---------------------------------------------------------------------------
__global__ void __launch_bounds__(256) stats_kernel(
    const float* __restrict__ gate, const int* __restrict__ block_ids,
    const unsigned char* __restrict__ pad,
    float* __restrict__ m_ws, float* __restrict__ s_ws, int* __restrict__ cnt_ws)
{
    __shared__ float gl[N_];
    __shared__ int   bl[N_];
    __shared__ unsigned char pl[N_];
    __shared__ int   mk[NB_];
    __shared__ float ss[NB_];
    __shared__ int   cc[NB_];
    const int b = blockIdx.x, tid = threadIdx.x;
#if HAVE_ASYNC_LDS
    for (int n = tid; n < N_; n += 256) {
        __builtin_amdgcn_global_load_async_to_lds_b32(
            (gint*)(gate + b * N_ + n), (sint*)&gl[n], 0, 0);
        __builtin_amdgcn_global_load_async_to_lds_b32(
            (gint*)(block_ids + b * N_ + n), (sint*)&bl[n], 0, 0);
        __builtin_amdgcn_global_load_async_to_lds_b8(
            (gchar*)(pad + b * N_ + n), (schar*)&pl[n], 0, 0);
    }
    async_wait0();
#else
    for (int n = tid; n < N_; n += 256) {
        gl[n] = gate[b * N_ + n];
        bl[n] = block_ids[b * N_ + n];
        pl[n] = pad[b * N_ + n];
    }
#endif
    if (tid < NB_) { mk[tid] = INT_MIN; ss[tid] = 0.f; cc[tid] = 0; }
    __syncthreads();
    for (int n = tid; n < N_; n += 256) {
        if (!pl[n]) {
            int id = bl[n];
            atomicMax(&mk[id], enc_f32(gl[n]));
            atomicAdd(&cc[id], 1);
        }
    }
    __syncthreads();
    for (int n = tid; n < N_; n += 256) {
        if (!pl[n]) {
            int id = bl[n];
            atomicAdd(&ss[id], __expf(gl[n] - dec_f32(mk[id])));
        }
    }
    __syncthreads();
    if (tid < NB_) {
        m_ws[b * NB_ + tid]   = (cc[tid] > 0) ? dec_f32(mk[tid]) : 0.f;
        s_ws[b * NB_ + tid]   = ss[tid];
        cnt_ws[b * NB_ + tid] = cc[tid];
    }
}

// ---------------------------------------------------------------------------
// Kernel 3: pooled[b,k,:] = sum_{n: bid==k} w[b,n] * x[b,n,:]
// One WG per (b,k); uniform branch on n; deterministic accumulation order.
// ---------------------------------------------------------------------------
__global__ void __launch_bounds__(256) pool_kernel(
    const float* __restrict__ x, const float* __restrict__ gate,
    const int* __restrict__ block_ids, const unsigned char* __restrict__ pad,
    const float* __restrict__ m_ws, const float* __restrict__ s_ws,
    const int* __restrict__ cnt_ws, const float* __restrict__ ct_emb,
    const int* __restrict__ cancer_type, float* __restrict__ pooled)
{
    __shared__ float gl[N_];
    __shared__ int   bl[N_];
    __shared__ unsigned char pl[N_];
    const int b = blockIdx.x >> 7;
    const int k = blockIdx.x & 127;
    const int tid = threadIdx.x;
#if HAVE_ASYNC_LDS
    for (int n = tid; n < N_; n += 256) {
        __builtin_amdgcn_global_load_async_to_lds_b32(
            (gint*)(gate + b * N_ + n), (sint*)&gl[n], 0, 0);
        __builtin_amdgcn_global_load_async_to_lds_b32(
            (gint*)(block_ids + b * N_ + n), (sint*)&bl[n], 0, 0);
        __builtin_amdgcn_global_load_async_to_lds_b8(
            (gchar*)(pad + b * N_ + n), (schar*)&pl[n], 0, 0);
    }
    async_wait0();
#else
    for (int n = tid; n < N_; n += 256) {
        gl[n] = gate[b * N_ + n];
        bl[n] = block_ids[b * N_ + n];
        pl[n] = pad[b * N_ + n];
    }
#endif
    __syncthreads();
    float4 acc = {0.f, 0.f, 0.f, 0.f};
    if (cnt_ws[b * NB_ + k] > 0) {
        const float m   = m_ws[b * NB_ + k];
        const float inv = 1.f / fmaxf(s_ws[b * NB_ + k], 1e-30f);
        const float ctf = 1.f + 0.1f * ct_emb[cancer_type[b] * NB_ + k];
        const float4* xr = (const float4*)x + (size_t)b * N_ * (H_ / 4);
        for (int n = 0; n < N_; ++n) {
            if (!pl[n] && bl[n] == k) {
                float wgt = __expf(gl[n] - m) * inv * ctf;
                float4 v = xr[(size_t)n * (H_ / 4) + tid];
                acc.x += wgt * v.x; acc.y += wgt * v.y;
                acc.z += wgt * v.z; acc.w += wgt * v.w;
            }
        }
    }
    ((float4*)pooled)[(size_t)blockIdx.x * (H_ / 4) + tid] = acc;
}

// ---------------------------------------------------------------------------
// Kernel 4: y = pooled @ Wp + bp   (WMMA GEMM, 1024x1024x1024)
// ---------------------------------------------------------------------------
__global__ void __launch_bounds__(256) proj_kernel(
    const float* __restrict__ pooled, const _Float16* __restrict__ wph,
    const float* __restrict__ bp, float* __restrict__ y)
{
    __shared__ _Float16 as16[16 * H_];   // 32KB
    const int tid  = threadIdx.x;
    const int lane = tid & 31;
    const int w    = tid >> 5;
    const int rt   = blockIdx.x;         // row tile (16 rows)
    const int cg   = blockIdx.y;         // col group (128 cols)

    const float4* pg = (const float4*)pooled + (size_t)rt * 16 * (H_ / 4);
    #pragma unroll
    for (int i = 0; i < 16; ++i) {
        int e4 = tid + i * 256;
        float4 v = pg[e4];
        v4h h; h[0] = (_Float16)v.x; h[1] = (_Float16)v.y;
               h[2] = (_Float16)v.z; h[3] = (_Float16)v.w;
        *(v4h*)&as16[e4 * 4] = h;
    }
    __syncthreads();

    const int colBase = cg * 128 + w * 16;
    const int arow = lane & 15;
    const int ahi  = (lane >> 4) & 1;
    const _Float16* xrow = &as16[arow * H_];

    v8f c = {};
    #pragma unroll 4
    for (int kt = 0; kt < 32; ++kt) {
        const _Float16* ap = xrow + kt * 32 + ahi * 8;
        v8h a0 = *(const v8h*)(ap);
        v8h a1 = *(const v8h*)(ap + 16);
        v16h a;
        #pragma unroll
        for (int e = 0; e < 8; ++e) { a[e] = a0[e]; a[e + 8] = a1[e]; }
        v16h bm;
        #pragma unroll
        for (int e = 0; e < 8; ++e) {
            bm[e]     = wph[(size_t)(kt * 32 + ahi * 8 + e) * H_ + colBase + arow];
            bm[e + 8] = wph[(size_t)(kt * 32 + 16 + ahi * 8 + e) * H_ + colBase + arow];
        }
        c = __builtin_amdgcn_wmma_f32_16x16x32_f16(
                false, a, false, bm, (short)0, c, false, false);
    }
    const int col = colBase + (lane & 15);
    const float bpc = bp[col];
    #pragma unroll
    for (int r = 0; r < 8; ++r) {
        int row = rt * 16 + r + ahi * 8;
        y[(size_t)row * H_ + col] = c[r] + bpc;
    }
}

// ---------------------------------------------------------------------------
// Kernel 5: LayerNorm + ELU + present-masking; writes block_tokens and has_b.
// ---------------------------------------------------------------------------
__global__ void __launch_bounds__(256) ln_kernel(
    const float* __restrict__ y, const float* __restrict__ ln_g,
    const float* __restrict__ ln_b, const int* __restrict__ cnt_ws,
    float* __restrict__ out, float* __restrict__ hasb_out)
{
    __shared__ float red[16];
    const int row = blockIdx.x;          // b*NB + k
    const int k   = row & 127;
    const int tid = threadIdx.x, lane = tid & 31, w = tid >> 5;

    float4 v = ((const float4*)y)[(size_t)row * (H_ / 4) + tid];
    float s1 = v.x + v.y + v.z + v.w;
    float s2 = v.x * v.x + v.y * v.y + v.z * v.z + v.w * v.w;
    #pragma unroll
    for (int m = 1; m < 32; m <<= 1) {
        s1 += __shfl_xor(s1, m, 32);
        s2 += __shfl_xor(s2, m, 32);
    }
    if (lane == 0) { red[w] = s1; red[8 + w] = s2; }
    __syncthreads();
    if (tid == 0) {
        float a = 0.f, q = 0.f;
        #pragma unroll
        for (int i = 0; i < 8; ++i) { a += red[i]; q += red[8 + i]; }
        red[0] = a; red[8] = q;
    }
    __syncthreads();
    const float mu  = red[0] * (1.f / H_);
    const float var = red[8] * (1.f / H_) - mu * mu;
    const float rs  = rsqrtf(var + LN_EPS);

    bool present = false;
    #pragma unroll
    for (int bb = 0; bb < B_; ++bb) present = present || (cnt_ws[bb * NB_ + k] > 0);

    const int col = tid * 4;
    float4 o;
    o.x = eluf((v.x - mu) * rs * ln_g[col + 0] + ln_b[col + 0]);
    o.y = eluf((v.y - mu) * rs * ln_g[col + 1] + ln_b[col + 1]);
    o.z = eluf((v.z - mu) * rs * ln_g[col + 2] + ln_b[col + 2]);
    o.w = eluf((v.w - mu) * rs * ln_g[col + 3] + ln_b[col + 3]);
    if (!present) { o.x = 0.f; o.y = 0.f; o.z = 0.f; o.w = 0.f; }
    ((float4*)out)[(size_t)row * (H_ / 4) + tid] = o;
    if (tid == 0) hasb_out[row] = (cnt_ws[row] > 0) ? 1.f : 0.f;
}

// ---------------------------------------------------------------------------
extern "C" void kernel_launch(void* const* d_in, const int* in_sizes, int n_in,
                              void* d_out, int out_size, void* d_ws, size_t ws_size,
                              hipStream_t stream)
{
    const float*         x   = (const float*)d_in[0];
    const int*           ct  = (const int*)d_in[1];
    const int*           bid = (const int*)d_in[2];
    const unsigned char* pad = (const unsigned char*)d_in[3];
    const float*         W1  = (const float*)d_in[4];
    const float*         b1  = (const float*)d_in[5];
    const float*         W2  = (const float*)d_in[6];
    const float*         b2  = (const float*)d_in[7];
    const float*         ctE = (const float*)d_in[8];
    const float*         Wp  = (const float*)d_in[9];
    const float*         bp  = (const float*)d_in[10];
    const float*         lg  = (const float*)d_in[11];
    const float*         lb  = (const float*)d_in[12];

    float* ws      = (float*)d_ws;
    float* gate_ws = ws;                          // 32768 f
    float* m_ws    = ws + 32768;                  // 1024 f
    float* s_ws    = m_ws + 1024;                 // 1024 f
    int*   cnt_ws  = (int*)(s_ws + 1024);         // 1024 i
    float* pooled  = s_ws + 2048;                 // 1024*1024 f
    float* y_ws    = pooled + (size_t)1024 * 1024;
    _Float16* w1h  = (_Float16*)(y_ws + (size_t)1024 * 1024);  // 512K f16
    _Float16* wph  = w1h + (size_t)HH_ * H_;                   // 1M f16

    float* out  = (float*)d_out;
    float* hasb = out + (size_t)B_ * NB_ * H_;

    cvt_f16_kernel<<<(H_ * HH_ / 4 + 255) / 256, 256, 0, stream>>>(W1, w1h, H_ * HH_ / 4);
    cvt_f16_kernel<<<(H_ * H_  / 4 + 255) / 256, 256, 0, stream>>>(Wp, wph, H_ * H_ / 4);

    size_t gateLds = (size_t)128 * H_ * sizeof(_Float16) + 1024 * sizeof(float);
    gate_kernel<<<256, 256, gateLds, stream>>>(x, w1h, b1, W2, b2, gate_ws);
    stats_kernel<<<B_, 256, 0, stream>>>(gate_ws, bid, pad, m_ws, s_ws, cnt_ws);
    pool_kernel<<<B_ * NB_, 256, 0, stream>>>(x, gate_ws, bid, pad, m_ws, s_ws,
                                              cnt_ws, ctE, ct, pooled);
    proj_kernel<<<dim3(64, 8), 256, 0, stream>>>(pooled, wph, bp, y_ws);
    ln_kernel<<<B_ * NB_, 256, 0, stream>>>(y_ws, lg, lb, cnt_ws, out, hasb);
}
